// FlashAttention_19112604467222
// MI455X (gfx1250) — matrix operand order repaired and checked by reading, not compiled, not run on hardware
//
#include <hip/hip_runtime.h>
#include <hip/hip_bf16.h>

// ---------------------------------------------------------------------------
// FlashAttention forward, B=4, S=4096, D=128, fp32 in/out.
// Compute-bound (AI ~1000 FLOP/B)  ->  v_wmma_f32_16x16x32_f16 for both GEMMs.
// Block = 128 threads (4 wave32), each wave owns a 16-row query tile.
// K/V tiles staged to LDS as f16 (V transposed); next tile prefetched
// (global_prefetch_b8) while the current tile's WMMAs run.
// ---------------------------------------------------------------------------

typedef _Float16 half_t;
typedef __attribute__((ext_vector_type(16))) _Float16 v16h;
typedef __attribute__((ext_vector_type(8)))  float    v8f;

#define THREADS 128
#define NW      4      // waves per block
#define QB      64     // query rows per block (16 per wave)
#define BK      64     // key tile
#define DHEAD   128
// LDS pitches (half_t elements): padded so transposed staging stores and the
// 16B fragment loads are bank-conflict-friendly; all pitches are multiples of
// 8 halves (16B) so b128 fragment loads stay aligned.
#define QP      136
#define KP      136
#define VP      72
#define PP      72

__device__ __forceinline__ v16h load_frag16(const half_t* p, int off_elems) {
  // two 16-byte LDS loads -> one 16xhalf fragment register set
  union { uint4 u[2]; v16h h; } x;
  x.u[0] = *reinterpret_cast<const uint4*>(p);
  x.u[1] = *reinterpret_cast<const uint4*>(p + off_elems);
  return x.h;
}

__global__ __launch_bounds__(THREADS)
void fa_fwd_wmma(const float* __restrict__ q,
                 const float* __restrict__ k,
                 const float* __restrict__ v,
                 float* __restrict__ out, int S) {
  __shared__ __align__(16) half_t Qs[QB * QP];        // [q_row][d]   (scaled)
  __shared__ __align__(16) half_t Ks[BK * KP];        // [key][d]     row-major
  __shared__ __align__(16) half_t Vt[DHEAD * VP];     // [d][key]     transposed
  __shared__ __align__(16) half_t Pw[NW * 16 * PP];   // per-wave P scratch [m][key]

  const int tid  = threadIdx.x;
  const int wid  = tid >> 5;
  const int lane = tid & 31;
  const int l16  = lane & 15;
  const int hi   = lane >> 4;          // 0: rows r, 1: rows r+8 in C layout
  const int b    = blockIdx.y;
  const int q0   = blockIdx.x * QB;

  const float scale = 0.08838834764831845f;   // 1/sqrt(128)

  // ---- stage Q tile once (fold in softmax scale) ----
  const float* qg = q + (size_t)b * S * DHEAD + (size_t)q0 * DHEAD;
  for (int i = tid * 4; i < QB * DHEAD; i += THREADS * 4) {
    int r = i >> 7, d = i & (DHEAD - 1);
    float4 qv = *reinterpret_cast<const float4*>(qg + i);
    half_t* qd = &Qs[r * QP + d];
    union { half_t h[4]; uint2 u; } pk;
    pk.h[0] = (half_t)(qv.x * scale);
    pk.h[1] = (half_t)(qv.y * scale);
    pk.h[2] = (half_t)(qv.z * scale);
    pk.h[3] = (half_t)(qv.w * scale);
    *reinterpret_cast<uint2*>(qd) = pk.u;
  }

  // ---- per-row running state (row m = r + 8*hi of this wave's 16-row tile) ----
  float m_i[8], l_i[8];
  v8f   o[8];
#pragma unroll
  for (int r = 0; r < 8; ++r) { m_i[r] = -3.0e38f; l_i[r] = 0.0f; }
#pragma unroll
  for (int n8 = 0; n8 < 8; ++n8) {
#pragma unroll
    for (int r = 0; r < 8; ++r) o[n8][r] = 0.0f;
  }

  const float* kg = k + (size_t)b * S * DHEAD;
  const float* vg = v + (size_t)b * S * DHEAD;
  half_t* pw = Pw + wid * 16 * PP;

  for (int kt = 0; kt < S; kt += BK) {
    __syncthreads();   // previous iteration done with Ks/Vt (covers Q stage on iter 0)

    // ---- stage K (row-major) and V (transposed) tiles as f16, float4 loads ----
    for (int i = tid * 4; i < BK * DHEAD; i += THREADS * 4) {
      int r = i >> 7, d = i & (DHEAD - 1);
      float4 kv = *reinterpret_cast<const float4*>(kg + (size_t)(kt + r) * DHEAD + d);
      float4 vv = *reinterpret_cast<const float4*>(vg + (size_t)(kt + r) * DHEAD + d);
      // K: 4 contiguous halves -> one 8-byte LDS store
      union { half_t h[4]; uint2 u; } pk;
      pk.h[0] = (half_t)kv.x; pk.h[1] = (half_t)kv.y;
      pk.h[2] = (half_t)kv.z; pk.h[3] = (half_t)kv.w;
      *reinterpret_cast<uint2*>(&Ks[r * KP + d]) = pk.u;
      // V: transposed scatter (pitch 72 halves keeps banks spread)
      Vt[(d + 0) * VP + r] = (half_t)vv.x;
      Vt[(d + 1) * VP + r] = (half_t)vv.y;
      Vt[(d + 2) * VP + r] = (half_t)vv.z;
      Vt[(d + 3) * VP + r] = (half_t)vv.w;
    }
    __syncthreads();

    // ---- prefetch next K/V tile into cache while we compute this one ----
    if (kt + BK < S) {
      const float* nk = kg + (size_t)(kt + BK) * DHEAD;
      const float* nv = vg + (size_t)(kt + BK) * DHEAD;
      // one 128B line per prefetch; BK*DHEAD*4B = 32KB each -> 256 lines each
      for (int off = tid * 32; off < BK * DHEAD; off += THREADS * 32) {
        __builtin_prefetch(nk + off, 0, 3);
        __builtin_prefetch(nv + off, 0, 3);
      }
    }

    // ---- S = (Q*scale) K^T : 4 N-tiles of 16, K-dim 128 in 4 chunks of 32 ----
    v8f s[4];
#pragma unroll
    for (int t = 0; t < 4; ++t)
#pragma unroll
      for (int r = 0; r < 8; ++r) s[t][r] = 0.0f;

#pragma unroll
    for (int c = 0; c < 4; ++c) {
      // A fragment: row m = wid*16 + l16; chunks at d = 32c + 8*hi and +16
      v16h a = load_frag16(Qs + (wid * 16 + l16) * QP + c * 32 + 8 * hi, 16);
#pragma unroll
      for (int t = 0; t < 4; ++t) {
        // B fragment (K^T): column n = key = t*16+l16; chunks at d = 32c + 8*hi and +16
        v16h bf = load_frag16(Ks + (t * 16 + l16) * KP + c * 32 + 8 * hi, 16);
        s[t] = __builtin_amdgcn_wmma_f32_16x16x32_f16(
            false, a, false, bf, (short)0, s[t], false, false);
      }
    }

    // ---- online softmax update (rows live on 16-lane halves) ----
    float alpha[8];
#pragma unroll
    for (int r = 0; r < 8; ++r) {
      float mt = fmaxf(fmaxf(s[0][r], s[1][r]), fmaxf(s[2][r], s[3][r]));
      mt = fmaxf(mt, __shfl_xor(mt, 1, 32));
      mt = fmaxf(mt, __shfl_xor(mt, 2, 32));
      mt = fmaxf(mt, __shfl_xor(mt, 4, 32));
      mt = fmaxf(mt, __shfl_xor(mt, 8, 32));
      float mnew = fmaxf(m_i[r], mt);
      alpha[r]   = __expf(m_i[r] - mnew);
      m_i[r]     = mnew;
      float sum = 0.0f;
#pragma unroll
      for (int t = 0; t < 4; ++t) {
        float p = __expf(s[t][r] - mnew);
        s[t][r] = p;
        sum += p;
      }
      sum += __shfl_xor(sum, 1, 32);
      sum += __shfl_xor(sum, 2, 32);
      sum += __shfl_xor(sum, 4, 32);
      sum += __shfl_xor(sum, 8, 32);
      l_i[r] = l_i[r] * alpha[r] + sum;
    }

    // ---- rescale O accumulators ----
#pragma unroll
    for (int n8 = 0; n8 < 8; ++n8)
#pragma unroll
      for (int r = 0; r < 8; ++r) o[n8][r] *= alpha[r];

    // ---- spill P (f16) to wave-local LDS: C layout -> row-major [m][key] ----
#pragma unroll
    for (int t = 0; t < 4; ++t)
#pragma unroll
      for (int r = 0; r < 8; ++r)
        pw[(r + 8 * hi) * PP + t * 16 + l16] = (half_t)s[t][r];
    // wave-local DS ops are in-order: RAW through LDS within a wave is safe.

    // ---- O += P V : 8 N-tiles over D, K-dim BK=64 in 2 chunks of 32 ----
#pragma unroll
    for (int c = 0; c < 2; ++c) {
      // A fragment of P: row m = l16
      v16h ap = load_frag16(pw + l16 * PP + c * 32 + 8 * hi, 16);
#pragma unroll
      for (int n8 = 0; n8 < 8; ++n8) {
        // B fragment of V: column n = d = n8*16+l16; key chunks at 32c + 8*hi and +16 via Vt
        v16h bf = load_frag16(Vt + (n8 * 16 + l16) * VP + c * 32 + 8 * hi, 16);
        o[n8] = __builtin_amdgcn_wmma_f32_16x16x32_f16(
            false, ap, false, bf, (short)0, o[n8], false, false);
      }
    }
  }

  // ---- epilogue: O /= l, write fp32 ----
  float* og = out + (size_t)b * S * DHEAD + (size_t)q0 * DHEAD;
#pragma unroll
  for (int n8 = 0; n8 < 8; ++n8) {
#pragma unroll
    for (int r = 0; r < 8; ++r) {
      int m = wid * 16 + r + 8 * hi;
      og[(size_t)m * DHEAD + n8 * 16 + l16] = o[n8][r] / l_i[r];
    }
  }
}

extern "C" void kernel_launch(void* const* d_in, const int* in_sizes, int n_in,
                              void* d_out, int out_size, void* d_ws, size_t ws_size,
                              hipStream_t stream) {
  const float* q = (const float*)d_in[0];
  const float* k = (const float*)d_in[1];
  const float* v = (const float*)d_in[2];
  float* out = (float*)d_out;

  const int B = 4;
  const int S = in_sizes[0] / (B * DHEAD);   // 4096

  dim3 grid(S / QB, B);   // (64, 4) = 256 workgroups
  dim3 block(THREADS);
  hipLaunchKernelGGL(fa_fwd_wmma, grid, block, 0, stream, q, k, v, out, S);
}
